// ResNetGNN_197568495815
// MI455X (gfx1250) — matrix operand order repaired and checked by reading, not compiled, not run on hardware
//
#include <hip/hip_runtime.h>

// ---------------------------------------------------------------------------
// CDNA5 / gfx1250 fused GNN pipeline. All GEMMs via v_wmma_f32_16x16x32_bf16.
// wave32, 256 threads/block = 8 waves.
//
// - Register-blocked GEMM: each wave owns a 16-row M block and a 16-column
//   phase, K outer loop, up to 6 accumulator tiles; A fragments loaded once
//   per K-step and reused across n-tiles.
// - Weights staged in LDS in B-fragment-swizzled order so every lane's 16
//   B values are 32 contiguous bytes (2 x ds_load_b128 per WMMA).
// - wave id forced uniform via readfirstlane so no exec-masking around WMMA.
// ---------------------------------------------------------------------------

typedef __bf16 bf16;
typedef __attribute__((ext_vector_type(16))) __bf16 v16bf;
typedef __attribute__((ext_vector_type(8)))  float  v8f;

#define TPB    256
#define NWAVES 8

// ---------------------------------------------------------------------------
// Register-blocked GEMM over one 64-row LDS tile:
//   D[m, n] = sum_k Xs[m, k] * Wz[k, n]   for this wave's (m0, n-phase) slice
// Xs: row-major [64 x K] bf16.
// Wz: B-swizzled weights: element (k, n) at Wz[((k>>5)*NW + n)*32 + (k&31)].
// A fragment (ISA 7.12.2, 16-bit A 16x32): lane L holds row M = L%16;
//   vector elems 0..7  = K[k0 + (L/16)*16 + 0..7]
//   vector elems 8..15 = K[k0 + (L/16)*16 + 8..15]  (contiguous k, same as B)
// B fragment (32x16): lane L holds col N = L%16, K run (L/16)*16 + 0..15,
//   which is exactly 32 contiguous bytes in the swizzled layout.
// C/D: vgpr v -> (M = (L/16)*8 + v, N = L%16).
// epi(m, n, val) invoked for every output element this wave owns.
// ---------------------------------------------------------------------------
template<int K, int NW, class EPI>
__device__ inline void lds_gemm(const bf16* Xs, const bf16* Wz,
                                int lane, int wave, EPI epi) {
  constexpr int NT = (NW + 31) / 32;   // max n-tiles per wave
  const int m0    = (wave & 3) * 16;   // this wave's M block (uniform)
  const int nbase = (wave >> 2) * 16;  // column phase: 0 or 16 (uniform)
  const int half  = lane >> 4;
  const int mrow  = m0 + (lane & 15);
  const int ncol0 = nbase + (lane & 15);

  v8f acc[NT];
  #pragma unroll
  for (int j = 0; j < NT; ++j) acc[j] = v8f{};

  #pragma unroll
  for (int k0 = 0; k0 < K; k0 += 32) {
    // A fragment: two contiguous 16-byte runs; loaded once per K-step.
    v16bf a;
    const bf16* ap = Xs + mrow * K + k0 + half * 16;
    #pragma unroll
    for (int i = 0; i < 8; ++i) a[i] = ap[i];
    #pragma unroll
    for (int i = 0; i < 8; ++i) a[8 + i] = ap[8 + i];

    #pragma unroll
    for (int j = 0; j < NT; ++j) {
      const int n0 = nbase + j * 32;
      if (NW % 32 == 0 || n0 < NW) {     // statically true when NW%32==0
        // B fragment: 32 contiguous bytes in the swizzled weight tile.
        const bf16* bp = Wz + (((unsigned)k0 >> 5) * NW + ncol0 + j * 32) * 32
                            + half * 16;
        v16bf b;
        #pragma unroll
        for (int i = 0; i < 16; ++i) b[i] = bp[i];
        acc[j] = __builtin_amdgcn_wmma_f32_16x16x32_bf16(
            /*neg_a=*/false, a, /*neg_b=*/false, b,
            /*c_mod=*/(short)0, acc[j], /*reuse_a=*/false, /*reuse_b=*/false);
      }
    }
  }

  #pragma unroll
  for (int j = 0; j < NT; ++j) {
    const int n0 = nbase + j * 32;
    if (NW % 32 == 0 || n0 < NW) {
      int n = n0 + (lane & 15);
      #pragma unroll
      for (int v = 0; v < 8; ++v) {
        int m = m0 + (lane >> 4) * 8 + v;
        epi(m, n, acc[j][v]);
      }
    }
  }
}

// Hidden layer: Hs = act(Xs @ Wz + bias), bf16 out (row-major [64 x NW]).
template<int K, int NW, bool RELU>
__device__ inline void lds_mlp_layer(const bf16* Xs, const bf16* Wz,
                                     const float* __restrict__ bias,
                                     bf16* Hs, int lane, int wave) {
  lds_gemm<K, NW>(Xs, Wz, lane, wave, [&](int m, int n, float val) {
    float h = val + bias[n];
    if (RELU) h = fmaxf(h, 0.f);
    Hs[m * NW + n] = (bf16)h;
  });
}

// Load a [kvalid x nvalid] row-major bf16 weight from global into a padded
// [Kp x Np] LDS tile in B-fragment-swizzled order:
//   element (k, n) -> Wz[((k>>5)*Np + n)*32 + (k&31)]
// (zeros outside the valid region). Kp must be a multiple of 32.
__device__ inline void load_w(const bf16* __restrict__ g, bf16* Wz,
                              int Kp, int Np, int kvalid, int nvalid, int tid) {
  for (int idx = tid; idx < Kp * Np; idx += TPB) {
    int k = idx / Np, n = idx % Np;
    bf16 v = (bf16)0.f;
    if (k < kvalid && n < nvalid) v = g[k * nvalid + n];
    Wz[(((unsigned)k >> 5) * Np + n) * 32 + (k & 31)] = v;
  }
}

// ---------------------------------------------------------------------------
__global__ void f32_to_bf16_kernel(const float* __restrict__ src,
                                   bf16* __restrict__ dst, int n) {
  int i = blockIdx.x * blockDim.x + threadIdx.x;
  if (i < n) dst[i] = (bf16)src[i];
}

__global__ void zero_f32_kernel(float* __restrict__ p, int n) {
  int i = blockIdx.x * blockDim.x + threadIdx.x;
  if (i < n) p[i] = 0.f;
}

// ---------------------------------------------------------------------------
// Node embedding: in-kernel feature build (nf + cell emb, padded to K=32),
// then 8->64->64->64 MLP. Writes node_f32 master + bf16 shadow.
// ---------------------------------------------------------------------------
__global__ void embed_nodes_kernel(const float* __restrict__ x,
                                   const float* __restrict__ a,
                                   const int* __restrict__ data_id_p,
                                   int N, int Aslots,
                                   const bf16* w0, const float* b0,
                                   const bf16* w1, const float* b1,
                                   const bf16* w2, const float* b2,
                                   float* __restrict__ node_f32,
                                   bf16* __restrict__ nodeb) {
  __shared__ bf16 Xs[64 * 32];
  __shared__ bf16 Ws[64 * 64];
  __shared__ bf16 Hs[64 * 64];
  __shared__ bf16 Hs2[64 * 64];
  const int tid  = threadIdx.x, lane = tid & 31;
  const int wave = __builtin_amdgcn_readfirstlane(tid >> 5);
  const int base = blockIdx.x * 64;

  if (tid < 64) {
    int i = base + tid;
    float f[8];
    if (i < N) {
      const float* xr = x + (size_t)i * 8;
      f[0] = xr[2]; f[1] = xr[3]; f[2] = xr[6]; f[3] = xr[7]; f[4] = xr[4];
      int tr = (int)xr[1];
      tr = min(max(tr, 0), Aslots - 1);
      int did = data_id_p[0];
      const float* cell = a + ((size_t)did * Aslots + tr) * 3;
      f[5] = cell[0]; f[6] = cell[1]; f[7] = cell[2];
    } else {
      for (int c = 0; c < 8; ++c) f[c] = 0.f;
    }
    for (int c = 0; c < 8; ++c)  Xs[tid * 32 + c] = (bf16)f[c];
    for (int c = 8; c < 32; ++c) Xs[tid * 32 + c] = (bf16)0.f;
  }
  load_w(w0, Ws, 32, 64, 8, 64, tid);
  __syncthreads();
  lds_mlp_layer<32, 64, true>(Xs, Ws, b0, Hs, lane, wave);
  __syncthreads();
  load_w(w1, Ws, 64, 64, 64, 64, tid);
  __syncthreads();
  lds_mlp_layer<64, 64, true>(Hs, Ws, b1, Hs2, lane, wave);
  __syncthreads();
  load_w(w2, Ws, 64, 64, 64, 64, tid);
  __syncthreads();
  lds_gemm<64, 64>(Hs2, Ws, lane, wave, [&](int m, int n, float val) {
    int i = base + m;
    if (i < N) {
      float o = val + b2[n];
      node_f32[(size_t)i * 64 + n] = o;
      nodeb[(size_t)i * 64 + n]    = (bf16)o;
    }
  });
}

// ---------------------------------------------------------------------------
// Edge embedding: ef = (d, xj-xi over cols {2,3,6,7}) padded to K=32, then
// 5->64->64->64 MLP -> edgeb (bf16).
// ---------------------------------------------------------------------------
__global__ void embed_edges_kernel(const float* __restrict__ x,
                                   const float* __restrict__ edge_attr,
                                   const int* __restrict__ ei,
                                   int N, int E,
                                   const bf16* w0, const float* b0,
                                   const bf16* w1, const float* b1,
                                   const bf16* w2, const float* b2,
                                   bf16* __restrict__ edgeb) {
  __shared__ bf16 Xs[64 * 32];
  __shared__ bf16 Ws[64 * 64];
  __shared__ bf16 Hs[64 * 64];
  __shared__ bf16 Hs2[64 * 64];
  const int tid  = threadIdx.x, lane = tid & 31;
  const int wave = __builtin_amdgcn_readfirstlane(tid >> 5);
  const int base = blockIdx.x * 64;

  if (tid < 64) {
    int e = base + tid;
    float f[5];
    if (e < E) {
      int s = ei[e], d = ei[E + e];
      f[0] = edge_attr[e];
      const int cm[4] = {2, 3, 6, 7};
      #pragma unroll
      for (int c = 0; c < 4; ++c)
        f[1 + c] = x[(size_t)s * 8 + cm[c]] - x[(size_t)d * 8 + cm[c]];
    } else {
      for (int c = 0; c < 5; ++c) f[c] = 0.f;
    }
    for (int c = 0; c < 5; ++c)  Xs[tid * 32 + c] = (bf16)f[c];
    for (int c = 5; c < 32; ++c) Xs[tid * 32 + c] = (bf16)0.f;
  }
  load_w(w0, Ws, 32, 64, 5, 64, tid);
  __syncthreads();
  lds_mlp_layer<32, 64, true>(Xs, Ws, b0, Hs, lane, wave);
  __syncthreads();
  load_w(w1, Ws, 64, 64, 64, 64, tid);
  __syncthreads();
  lds_mlp_layer<64, 64, true>(Hs, Ws, b1, Hs2, lane, wave);
  __syncthreads();
  load_w(w2, Ws, 64, 64, 64, 64, tid);
  __syncthreads();
  lds_gemm<64, 64>(Hs2, Ws, lane, wave, [&](int m, int n, float val) {
    int e = base + m;
    if (e < E) edgeb[(size_t)e * 64 + n] = (bf16)(val + b2[n]);
  });
}

// ---------------------------------------------------------------------------
// Message-passing edge kernel (lin_edge 192->192->192->64, fused):
// gathers [edge | node[dst] | node[src]] into LDS, runs 3-layer MLP via WMMA,
// scatter-adds msg into aggr (f32 atomics), and does edge += msg in place.
// Activation ping-pong: Xs -> Hs -> Xs.
// ---------------------------------------------------------------------------
__global__ void mp_edge_kernel(const bf16* __restrict__ nodeb,
                               const int* __restrict__ ei,
                               int N, int E,
                               const bf16* w0, const float* b0,
                               const bf16* w1, const float* b1,
                               const bf16* w2, const float* b2,
                               bf16* __restrict__ edgeb,
                               float* __restrict__ aggr) {
  __shared__ bf16 Xs[64 * 192];
  __shared__ bf16 Hs[64 * 192];
  __shared__ bf16 Ws[192 * 192];
  __shared__ int  dsts[64];
  __shared__ int  srcs[64];
  const int tid  = threadIdx.x, lane = tid & 31;
  const int wave = __builtin_amdgcn_readfirstlane(tid >> 5);
  const int base = blockIdx.x * 64;

  if (tid < 64) {
    int e = base + tid;
    dsts[tid] = (e < E) ? ei[E + e] : 0;
    srcs[tid] = (e < E) ? ei[e] : 0;
  }
  __syncthreads();
  for (int idx = tid; idx < 64 * 192; idx += TPB) {
    int r = idx / 192, c = idx % 192;
    int e = base + r;
    bf16 v = (bf16)0.f;
    if (e < E) {
      if (c < 64)       v = edgeb[(size_t)e * 64 + c];
      else if (c < 128) v = nodeb[(size_t)dsts[r] * 64 + (c - 64)];
      else              v = nodeb[(size_t)srcs[r] * 64 + (c - 128)];
    }
    Xs[idx] = v;
  }
  load_w(w0, Ws, 192, 192, 192, 192, tid);
  __syncthreads();
  lds_mlp_layer<192, 192, true>(Xs, Ws, b0, Hs, lane, wave);
  __syncthreads();
  load_w(w1, Ws, 192, 192, 192, 192, tid);
  __syncthreads();
  lds_mlp_layer<192, 192, true>(Hs, Ws, b1, Xs, lane, wave);
  __syncthreads();
  load_w(w2, Ws, 192, 64, 192, 64, tid);
  __syncthreads();
  lds_gemm<192, 64>(Xs, Ws, lane, wave, [&](int m, int n, float val) {
    int e = base + m;
    if (e < E) {
      float msg = val + b2[n];
      atomicAdd(&aggr[(size_t)dsts[m] * 64 + n], msg);
      size_t ix = (size_t)e * 64 + n;
      edgeb[ix] = (bf16)((float)edgeb[ix] + msg);
    }
  });
}

// ---------------------------------------------------------------------------
// Node update (lin_node 128->128->128->64, fused) with residual into the f32
// master node buffer; refreshes the bf16 shadow. Ping-pong Xs -> Hs -> Xs.
// ---------------------------------------------------------------------------
__global__ void mp_node_kernel(const float* __restrict__ aggr, int N,
                               const bf16* w0, const float* b0,
                               const bf16* w1, const float* b1,
                               const bf16* w2, const float* b2,
                               float* __restrict__ node_f32,
                               bf16* __restrict__ nodeb) {
  __shared__ bf16 Xs[64 * 128];
  __shared__ bf16 Hs[64 * 128];
  __shared__ bf16 Ws[128 * 128];
  const int tid  = threadIdx.x, lane = tid & 31;
  const int wave = __builtin_amdgcn_readfirstlane(tid >> 5);
  const int base = blockIdx.x * 64;

  for (int idx = tid; idx < 64 * 128; idx += TPB) {
    int r = idx / 128, c = idx % 128;
    int i = base + r;
    bf16 v = (bf16)0.f;
    if (i < N)
      v = (c < 64) ? nodeb[(size_t)i * 64 + c]
                   : (bf16)aggr[(size_t)i * 64 + (c - 64)];
    Xs[idx] = v;
  }
  load_w(w0, Ws, 128, 128, 128, 128, tid);
  __syncthreads();
  lds_mlp_layer<128, 128, true>(Xs, Ws, b0, Hs, lane, wave);
  __syncthreads();
  load_w(w1, Ws, 128, 128, 128, 128, tid);
  __syncthreads();
  lds_mlp_layer<128, 128, true>(Hs, Ws, b1, Xs, lane, wave);
  __syncthreads();
  load_w(w2, Ws, 128, 64, 128, 64, tid);
  __syncthreads();
  lds_gemm<128, 64>(Xs, Ws, lane, wave, [&](int m, int n, float val) {
    int i = base + m;
    if (i < N) {
      size_t ix = (size_t)i * 64 + n;
      float nv = node_f32[ix] + val + b2[n];
      node_f32[ix] = nv;
      nodeb[ix]    = (bf16)nv;
    }
  });
}

// ---------------------------------------------------------------------------
// Output head: 64->128->128->1. Final layer padded to N=16, col 0 kept.
// ---------------------------------------------------------------------------
__global__ void head_kernel(const bf16* __restrict__ nodeb, int N,
                            const bf16* w0, const float* b0,
                            const bf16* w1, const float* b1,
                            const bf16* w2, const float* b2,
                            float* __restrict__ out) {
  __shared__ bf16 Xs[64 * 64];
  __shared__ bf16 Hs[64 * 128];
  __shared__ bf16 Hs2[64 * 128];
  __shared__ bf16 Ws[128 * 128];
  const int tid  = threadIdx.x, lane = tid & 31;
  const int wave = __builtin_amdgcn_readfirstlane(tid >> 5);
  const int base = blockIdx.x * 64;

  for (int idx = tid; idx < 64 * 64; idx += TPB) {
    int r = idx / 64, c = idx % 64;
    int i = base + r;
    Xs[idx] = (i < N) ? nodeb[(size_t)i * 64 + c] : (bf16)0.f;
  }
  load_w(w0, Ws, 64, 128, 64, 128, tid);
  __syncthreads();
  lds_mlp_layer<64, 128, true>(Xs, Ws, b0, Hs, lane, wave);
  __syncthreads();
  load_w(w1, Ws, 128, 128, 128, 128, tid);
  __syncthreads();
  lds_mlp_layer<128, 128, true>(Hs, Ws, b1, Hs2, lane, wave);
  __syncthreads();
  load_w(w2, Ws, 128, 16, 128, 1, tid);  // [128,1] padded to [128,16]
  __syncthreads();
  lds_gemm<128, 16>(Hs2, Ws, lane, wave, [&](int m, int n, float val) {
    int i = base + m;
    if (n == 0 && i < N) out[i] = val + b2[0];
  });
}

// ---------------------------------------------------------------------------
extern "C" void kernel_launch(void* const* d_in, const int* in_sizes, int n_in,
                              void* d_out, int out_size, void* d_ws,
                              size_t ws_size, hipStream_t stream) {
  const float* x   = (const float*)d_in[0];
  const int*   ei  = (const int*)d_in[1];
  const float* ea  = (const float*)d_in[2];
  const int*   did = (const int*)d_in[3];
  const float* a   = (const float*)d_in[4];

  const int N      = in_sizes[0] / 8;
  const int E      = in_sizes[2];
  const int Aslots = in_sizes[4] / 9;  // a: [3, N+1, 3]

  const float* en_b[3] = {(const float*)d_in[6],  (const float*)d_in[8],  (const float*)d_in[10]};
  const float* ee_b[3] = {(const float*)d_in[12], (const float*)d_in[14], (const float*)d_in[16]};
  const float* le_b[3] = {(const float*)d_in[18], (const float*)d_in[20], (const float*)d_in[22]};
  const float* ln_b[3] = {(const float*)d_in[24], (const float*)d_in[26], (const float*)d_in[28]};
  const float* no_b[3] = {(const float*)d_in[30], (const float*)d_in[32], (const float*)d_in[34]};

  // Workspace carve (256B aligned)
  char* basep = (char*)d_ws;
  size_t off = 0;
  auto carve = [&](size_t bytes) -> char* {
    off = (off + 255) & ~(size_t)255;
    char* r = basep + off;
    off += bytes;
    return r;
  };
  float* node_f32 = (float*)carve((size_t)N * 64 * 4);
  float* aggr     = (float*)carve((size_t)N * 64 * 4);
  bf16*  nodeb    = (bf16*)carve((size_t)N * 64 * 2);
  bf16*  edgeb    = (bf16*)carve((size_t)E * 64 * 2);

  // bf16 weight copies: en W0-2, ee W0-2, le W0-2 (both layers), ln W0-2, no W0-2
  const int widx[15] = {5, 7, 9, 11, 13, 15, 17, 19, 21, 23, 25, 27, 29, 31, 33};
  bf16* wb[15];
  for (int k = 0; k < 15; ++k) {
    int n = in_sizes[widx[k]];
    wb[k] = (bf16*)carve((size_t)n * 2);
    f32_to_bf16_kernel<<<(n + TPB - 1) / TPB, TPB, 0, stream>>>(
        (const float*)d_in[widx[k]], wb[k], n);
  }

  const int nblk = (N + 63) / 64;
  const int eblk = (E + 63) / 64;

  embed_nodes_kernel<<<nblk, TPB, 0, stream>>>(
      x, a, did, N, Aslots,
      wb[0], en_b[0], wb[1], en_b[1], wb[2], en_b[2], node_f32, nodeb);

  embed_edges_kernel<<<eblk, TPB, 0, stream>>>(
      x, ea, ei, N, E,
      wb[3], ee_b[0], wb[4], ee_b[1], wb[5], ee_b[2], edgeb);

  for (int l = 0; l < 2; ++l) {
    zero_f32_kernel<<<((N * 64) + TPB - 1) / TPB, TPB, 0, stream>>>(aggr, N * 64);
    mp_edge_kernel<<<eblk, TPB, 0, stream>>>(
        nodeb, ei, N, E,
        wb[6] + (size_t)l * 192 * 192, le_b[0] + (size_t)l * 192,
        wb[7] + (size_t)l * 192 * 192, le_b[1] + (size_t)l * 192,
        wb[8] + (size_t)l * 192 * 64,  le_b[2] + (size_t)l * 64,
        edgeb, aggr);
    mp_node_kernel<<<nblk, TPB, 0, stream>>>(
        aggr, N,
        wb[9]  + (size_t)l * 128 * 128, ln_b[0] + (size_t)l * 128,
        wb[10] + (size_t)l * 128 * 128, ln_b[1] + (size_t)l * 128,
        wb[11] + (size_t)l * 128 * 64,  ln_b[2] + (size_t)l * 64,
        node_f32, nodeb);
  }

  head_kernel<<<nblk, TPB, 0, stream>>>(
      nodeb, N, wb[12], no_b[0], wb[13], no_b[1], wb[14], no_b[2],
      (float*)d_out);

  (void)n_in; (void)out_size; (void)ws_size;
}
